// FineGenerator_17454747091676
// MI455X (gfx1250) — matrix operand order repaired and checked by reading, not compiled, not run on hardware
//
#include <hip/hip_runtime.h>
#include <math.h>

typedef __attribute__((ext_vector_type(16))) _Float16 v16h;
typedef __attribute__((ext_vector_type(8)))  _Float16 v8h;
typedef __attribute__((ext_vector_type(8)))  float    v8f;

// ---------------------------------------------------------------------------
// Implicit-GEMM convolution with WMMA f16->f32.
//   Activations are NHWC f16 with channel stride Cs = align32(cin), pad
//   channels kept zero by producers -> staged runs never cross a channel
//   boundary: A-stage is branch-light (vector loads or vector zeros).
//   k = (ky*kk+kx)*Cs + ci ; tap-nested K loop -> no division in hot loop.
//   B[k][n] pre-packed as Bt[n][k] f16, zero padded.
//   out: fp32 [M][Nout] (conv output, pre-activation, no bias).
// Block: 128 threads = 4 waves stacked in M; block tile 128(M) x 32(N);
// each wave owns a 32x32 tile: 2 A-frags x 2 B-frags -> 4 WMMAs per K-step
// (8 fragment ds_loads per 4 WMMAs; 4 WMMAs per barrier pair).
// ---------------------------------------------------------------------------
__global__ __launch_bounds__(128)
void k_gemm(const _Float16* __restrict__ act, const _Float16* __restrict__ Bt,
            float* __restrict__ out,
            int Hin, int Win, int Cs,
            int Hout, int Wout, int Mtot, int Nout,
            int Kpad, int kk, int stride, int pad, int dil)
{
    __shared__ __align__(16) _Float16 As[128][32];
    __shared__ __align__(16) _Float16 Bs[32][32];

    const int tid = threadIdx.x;
    const int lid = tid & 31;
    const int wv  = tid >> 5;
    const int mo  = wv * 32;            // wave M offset in block tile
    const int l15 = lid & 15;
    const int hfl = lid >> 4;           // lane half (0: lanes 0-15, 1: 16-31)
    const int mBase = blockIdx.x * 128;
    const int nBase = blockIdx.y * 32;

    // ---- A staging: thread owns full row As[tid][0..31] (64B) ----
    // ---- B staging: thread owns Bs[brow][bkg .. bkg+8) ----
    const int brow = tid >> 2;          // 0..31
    const int bkg  = (tid & 3) * 8;

    // one-time spatial decode of this thread's staged A row
    const int  am   = mBase + tid;
    const bool mval = (am < Mtot);
    const int  mc   = mval ? am : (Mtot - 1);
    const int  ox   = mc % Wout;
    const int  tm   = mc / Wout;
    const int  oy   = tm % Hout;
    const int  bb   = tm / Hout;
    const int  iy0  = oy * stride - pad;   // + dil*ky per tap
    const int  ix0  = ox * stride - pad;   // + dil*kx per tap
    const long rowb = (long)bb * Hin;      // batch row base (in lines)

    const _Float16* btrow = Bt + (long)(nBase + brow) * Kpad + bkg;

    v8f acc00 = {}, acc01 = {}, acc10 = {}, acc11 = {};
    const v8h zero8 = {};

    for (int ky = 0; ky < kk; ++ky) {
        const int  iy = iy0 + dil * ky;
        const bool yv = mval && iy >= 0 && iy < Hin;
        for (int kx = 0; kx < kk; ++kx) {
            const int  ix = ix0 + dil * kx;
            const bool sv = yv && ix >= 0 && ix < Win;
            const _Float16* arowp = act + ((rowb + iy) * Win + ix) * Cs;
            for (int c0 = 0; c0 < Cs; c0 += 32) {
                // ---- stage A: 64B row (4x16B) or zeros ----
                if (sv) {
                    const _Float16* src = arowp + c0;
                    *(v8h*)&As[tid][0]  = *(const v8h*)(src);
                    *(v8h*)&As[tid][8]  = *(const v8h*)(src + 8);
                    *(v8h*)&As[tid][16] = *(const v8h*)(src + 16);
                    *(v8h*)&As[tid][24] = *(const v8h*)(src + 24);
                } else {
                    *(v8h*)&As[tid][0]  = zero8;
                    *(v8h*)&As[tid][8]  = zero8;
                    *(v8h*)&As[tid][16] = zero8;
                    *(v8h*)&As[tid][24] = zero8;
                }
                // ---- stage B (pre-padded: unconditional 16B load) ----
                *(v8h*)&Bs[brow][bkg] = *(const v8h*)btrow;
                btrow += 32;
                __syncthreads();

                // ---- fragments per CDNA5 WMMA layouts ----
                // A 16x32: lanes 0-15 K {0..7,16..23}; lanes 16-31 {8..15,24..31}
                v8h a00 = *(const v8h*)&As[mo + l15][hfl * 8];
                v8h a01 = *(const v8h*)&As[mo + l15][16 + hfl * 8];
                v8h a10 = *(const v8h*)&As[mo + 16 + l15][hfl * 8];
                v8h a11 = *(const v8h*)&As[mo + 16 + l15][16 + hfl * 8];
                // B 32x16: column = lane&15; lanes 0-15 K {0..7,16..23}; lanes 16-31 {8..15,24..31}
                v8h b00 = *(const v8h*)&Bs[l15][hfl * 8];
                v8h b01 = *(const v8h*)&Bs[l15][16 + hfl * 8];
                v8h b10 = *(const v8h*)&Bs[16 + l15][hfl * 8];
                v8h b11 = *(const v8h*)&Bs[16 + l15][16 + hfl * 8];
                v16h af0, af1, bf0, bf1;
#pragma unroll
                for (int j = 0; j < 8; ++j) {
                    af0[j] = a00[j]; af0[8 + j] = a01[j];
                    af1[j] = a10[j]; af1[8 + j] = a11[j];
                    bf0[j] = b00[j]; bf0[8 + j] = b01[j];
                    bf1[j] = b10[j]; bf1[8 + j] = b11[j];
                }
                acc00 = __builtin_amdgcn_wmma_f32_16x16x32_f16(
                            false, af0, false, bf0, (short)0, acc00, false, false);
                acc01 = __builtin_amdgcn_wmma_f32_16x16x32_f16(
                            false, af0, false, bf1, (short)0, acc01, false, false);
                acc10 = __builtin_amdgcn_wmma_f32_16x16x32_f16(
                            false, af1, false, bf0, (short)0, acc10, false, false);
                acc11 = __builtin_amdgcn_wmma_f32_16x16x32_f16(
                            false, af1, false, bf1, (short)0, acc11, false, false);
                __syncthreads();
            }
        }
    }

    // ---- epilogue: C/D layout row M = g + 8*half, col N = lane&15 ----
    const int n0 = nBase + l15;
    const int n1 = nBase + 16 + l15;
#pragma unroll
    for (int g = 0; g < 8; ++g) {
        const int m0 = mBase + mo + g + hfl * 8;
        const int m1 = m0 + 16;
        if (m0 < Mtot) {
            if (n0 < Nout) out[(long)m0 * Nout + n0] = acc00[g];
            if (n1 < Nout) out[(long)m0 * Nout + n1] = acc01[g];
        }
        if (m1 < Mtot) {
            if (n0 < Nout) out[(long)m1 * Nout + n0] = acc10[g];
            if (n1 < Nout) out[(long)m1 * Nout + n1] = acc11[g];
        }
    }
}

// Pack OIHW fp32 weights -> Bt[n][tap*cpad + ci] f16, zero padded to Npad/Kpad.
__global__ void k_pack_w(const float* __restrict__ w, _Float16* __restrict__ Bt,
                         int cout, int cin, int kk, int cpad, int Kpad, long tot)
{
    long idx = (long)blockIdx.x * blockDim.x + threadIdx.x;
    if (idx >= tot) return;
    int n = (int)(idx / Kpad);
    int k = (int)(idx - (long)n * Kpad);
    int tap = k / cpad;
    int ci  = k - tap * cpad;
    float v = 0.f;
    if (n < cout && ci < cin) {
        int ky = tap / kk, kx = tap - ky * kk;
        v = w[(((long)n * cin + ci) * kk + ky) * kk + kx];
    }
    Bt[idx] = (_Float16)v;
}

// Gated activation: y = conv + bias; out = act(y[:cnum]) * sigmoid(y[cnum:]).
// Writes cOutPad channels per pixel (pad channels zeroed) at outOff in a
// buffer with channel stride outStride.
__global__ void k_gated_act(const float* __restrict__ conv,
                            const float* __restrict__ bias,
                            _Float16* __restrict__ out,
                            long M, int cnum, int convStride,
                            int cOutPad, int outStride, int outOff, int relu)
{
    long idx = (long)blockIdx.x * blockDim.x + threadIdx.x;
    if (idx >= M * cOutPad) return;
    long m = idx / cOutPad;
    int  c = (int)(idx - m * cOutPad);
    float o = 0.f;
    if (c < cnum) {
        float a = conv[m * convStride + c] + bias[c];
        float g = conv[m * convStride + c + cnum] + bias[c + cnum];
        float av = relu ? fmaxf(a, 0.f) : (a > 0.f ? a : (expf(a) - 1.f));
        o = av / (1.f + expf(-g));
    }
    out[m * outStride + outOff + c] = (_Float16)o;
}

// NCHW fp32 -> NHWC f16 with zero-padded channel stride Cs.
__global__ void k_nchw_to_nhwc(const float* __restrict__ in,
                               _Float16* __restrict__ out,
                               int Bn, int C, int Cs, int H, int W)
{
    long idx = (long)blockIdx.x * blockDim.x + threadIdx.x;
    long tot = (long)Bn * H * W * Cs;
    if (idx >= tot) return;
    int c = (int)(idx % Cs); long t = idx / Cs;
    int x = (int)(t % W); t /= W;
    int y = (int)(t % H); int b = (int)(t / H);
    float v = (c < C) ? in[(((long)b * C + c) * H + y) * W + x] : 0.f;
    out[idx] = (_Float16)v;
}

__global__ void k_up2x(const _Float16* __restrict__ in, _Float16* __restrict__ out,
                       int Bn, int H, int W, int C)
{
    long idx = (long)blockIdx.x * blockDim.x + threadIdx.x;
    long tot = (long)Bn * (2 * H) * (2 * W) * C;
    if (idx >= tot) return;
    int c = (int)(idx % C); long t = idx / C;
    int x = (int)(t % (2 * W)); t /= (2 * W);
    int y = (int)(t % (2 * H)); int b = (int)(t / (2 * H));
    out[idx] = in[((long)((b * H + (y >> 1)) * W + (x >> 1))) * C + c];
}

__global__ void k_down2(const _Float16* __restrict__ in, _Float16* __restrict__ out,
                        int Bn, int H, int W, int C)
{
    int Ho = H >> 1, Wo = W >> 1;
    long idx = (long)blockIdx.x * blockDim.x + threadIdx.x;
    long tot = (long)Bn * Ho * Wo * C;
    if (idx >= tot) return;
    int c = (int)(idx % C); long t = idx / C;
    int x = (int)(t % Wo); t /= Wo;
    int y = (int)(t % Ho); int b = (int)(t / Ho);
    out[idx] = in[((long)((b * H + 2 * y) * W + 2 * x)) * C + c];
}

// Extract 3x3 patches of bd, L2-normalize, store as GEMM weights
// Bt[b*L + l][(ky*3+kx)*C + ci]. One wave per (b,l).
__global__ void k_ca_wn(const _Float16* __restrict__ bd, _Float16* __restrict__ Bt,
                        int hb, int wb, int C, int Kp)
{
    int bl = blockIdx.x;
    int L  = hb * wb;
    int b  = bl / L, l = bl - b * L;
    int py = l / wb, px = l - py * wb;
    int lane = threadIdx.x;
    float sum = 0.f;
    for (int e = lane; e < Kp; e += 32) {
        int tap = e / C, ci = e - tap * C;
        int ky = tap / 3, kx = tap - ky * 3;
        int iy = py - 1 + ky, ix = px - 1 + kx;
        float v = 0.f;
        if (iy >= 0 && iy < hb && ix >= 0 && ix < wb)
            v = (float)bd[((long)((b * hb + iy) * wb + ix)) * C + ci];
        sum += v * v;
    }
#pragma unroll
    for (int o = 16; o > 0; o >>= 1) sum += __shfl_down(sum, o, 32);
    float nrm = fmaxf(sqrtf(__shfl(sum, 0, 32)), 1e-4f);
    for (int e = lane; e < Kp; e += 32) {
        int tap = e / C, ci = e - tap * C;
        int ky = tap / 3, kx = tap - ky * 3;
        int iy = py - 1 + ky, ix = px - 1 + kx;
        float v = 0.f;
        if (iy >= 0 && iy < hb && ix >= 0 && ix < wb)
            v = (float)bd[((long)((b * hb + iy) * wb + ix)) * C + ci];
        Bt[(long)bl * Kp + e] = (_Float16)(v / nrm);
    }
}

// mm[b][l] = 1 if 3x3 box-sum of (mask[::8,::8] > 0) is zero, else 0.
__global__ void k_mask(const float* __restrict__ mask, float* __restrict__ mm,
                       int Bn, int hb, int wb, int Hm, int Wm, int step)
{
    long idx = (long)blockIdx.x * blockDim.x + threadIdx.x;
    long tot = (long)Bn * hb * wb;
    if (idx >= tot) return;
    int l = (int)(idx % (hb * wb));
    int b = (int)(idx / (hb * wb));
    int y = l / wb, x = l - y * wb;
    float s = 0.f;
    for (int dy = -1; dy <= 1; ++dy)
        for (int dx = -1; dx <= 1; ++dx) {
            int yy = y + dy, xx = x + dx;
            if (yy >= 0 && yy < hb && xx >= 0 && xx < wb)
                s += (mask[(long)b * Hm * Wm + (yy * step) * Wm + xx * step] > 0.f)
                         ? 1.f : 0.f;
        }
    mm[idx] = (s == 0.f) ? 1.f : 0.f;
}

// fuse pass 1: f1[b][i=l][j=pix] = sum_d y[i+d][j+d], y[i][j] = corr[b][j][i]
__global__ void k_fuse1(const float* __restrict__ corr, float* __restrict__ f1)
{
    long idx = (long)blockIdx.x * blockDim.x + threadIdx.x;
    if (idx >= (long)4 * 1024 * 1024) return;
    int j = (int)(idx & 1023);
    long t = idx >> 10;
    int i = (int)(t & 1023);
    int b = (int)(t >> 10);
    float s = 0.f;
#pragma unroll
    for (int d = -1; d <= 1; ++d) {
        int ii = i + d, jj = j + d;
        if (ii >= 0 && ii < 1024 && jj >= 0 && jj < 1024)
            s += corr[((long)b * 1024 + jj) * 1024 + ii];
    }
    f1[idx] = s;
}

// fuse pass 2 in the (transpose(1,0,3,2)) view; output f2[b][pix][l].
// swap(v) = (v%32)*32 + v/32 is the involutive 32x32 index transpose.
__global__ void k_fuse2(const float* __restrict__ f1, float* __restrict__ f2)
{
    long idx = (long)blockIdx.x * blockDim.x + threadIdx.x;
    if (idx >= (long)4 * 1024 * 1024) return;
    int l = (int)(idx & 1023);
    long t = idx >> 10;
    int pix = (int)(t & 1023);
    int b   = (int)(t >> 10);
    int i2 = (l & 31) * 32 + (l >> 5);
    int j2 = (pix & 31) * 32 + (pix >> 5);
    float s = 0.f;
#pragma unroll
    for (int d = -1; d <= 1; ++d) {
        int ii = i2 + d, jj = j2 + d;
        if (ii >= 0 && ii < 1024 && jj >= 0 && jj < 1024) {
            int ls = (ii & 31) * 32 + (ii >> 5);
            int ps = (jj & 31) * 32 + (jj >> 5);
            s += f1[((long)b * 1024 + ls) * 1024 + ps];
        }
    }
    f2[idx] = s;
}

// masked softmax over L=1024 patches per (b,pix); one 128-thread block each.
__global__ __launch_bounds__(128)
void k_softmax(const float* __restrict__ f2, const float* __restrict__ mm,
               float* __restrict__ attn)
{
    __shared__ float red[128];
    const int bp = blockIdx.x;        // b*1024 + pix
    const int b  = bp >> 10;
    const float* src = f2 + (long)bp * 1024;
    const float* mb  = mm + b * 1024;
    float mx = -1e30f;
    for (int l = threadIdx.x; l < 1024; l += 128)
        mx = fmaxf(mx, src[l] * mb[l]);
    red[threadIdx.x] = mx; __syncthreads();
    for (int o = 64; o > 0; o >>= 1) {
        if (threadIdx.x < o)
            red[threadIdx.x] = fmaxf(red[threadIdx.x], red[threadIdx.x + o]);
        __syncthreads();
    }
    mx = red[0]; __syncthreads();
    float s = 0.f;
    for (int l = threadIdx.x; l < 1024; l += 128)
        s += expf(src[l] * mb[l] - mx);
    red[threadIdx.x] = s; __syncthreads();
    for (int o = 64; o > 0; o >>= 1) {
        if (threadIdx.x < o) red[threadIdx.x] += red[threadIdx.x + o];
        __syncthreads();
    }
    float inv = 1.f / red[0];
    float* dst = attn + (long)bp * 1024;
    for (int l = threadIdx.x; l < 1024; l += 128)
        dst[l] = expf(src[l] * mb[l] - mx) * inv * mb[l];
}

// Raw 4x4 stride-2 pad-1 patches of x2: rw[b][ky*4+kx][l][c]
__global__ void k_raww(const _Float16* __restrict__ x2, _Float16* __restrict__ rw,
                       int Hf, int C)
{
    long idx = (long)blockIdx.x * blockDim.x + threadIdx.x;
    long tot = (long)4 * 16 * 1024 * C;
    if (idx >= tot) return;
    int c = (int)(idx % C); long t = idx / C;
    int l = (int)(t & 1023); t >>= 10;
    int tap = (int)(t & 15);
    int b   = (int)(t >> 4);
    int ky = tap >> 2, kx = tap & 3;
    int py = l >> 5,  px = l & 31;
    int iy = py * 2 - 1 + ky, ix = px * 2 - 1 + kx;
    _Float16 v = (_Float16)0.f;
    if (iy >= 0 && iy < Hf && ix >= 0 && ix < Hf)
        v = x2[((long)((b * Hf + iy) * Hf) + ix) * C + c];
    rw[idx] = v;
}

// Transposed conv (lhs_dilation=2, pad=2, k=4) of attn with flipped raw
// patches; out = 0.25 * result, written NHWC f16 (96 ch @ 64x64).
__global__ void k_deconv(const float* __restrict__ attn,
                         const _Float16* __restrict__ rw,
                         _Float16* __restrict__ out, int C)
{
    long idx = (long)blockIdx.x * blockDim.x + threadIdx.x;
    long tot = (long)4 * 64 * 64 * C;
    if (idx >= tot) return;
    int c = (int)(idx % C); long t = idx / C;
    int x = (int)(t & 63); t >>= 6;
    int y = (int)(t & 63);
    int b = (int)(t >> 6);
    float acc = 0.f;
    for (int ky = (y & 1); ky < 4; ky += 2) {
        int iy = (y + ky - 2) >> 1;
        if (iy < 0 || iy >= 32) continue;
        for (int kx = (x & 1); kx < 4; kx += 2) {
            int ix = (x + kx - 2) >> 1;
            if (ix < 0 || ix >= 32) continue;
            const float* ap = attn + ((long)(b * 1024) + iy * 32 + ix) * 1024;
            const _Float16* wp =
                rw + ((long)(b * 16 + (3 - ky) * 4 + (3 - kx)) * 1024) * C + c;
            for (int l = 0; l < 1024; ++l)
                acc += ap[l] * (float)wp[(long)l * C];
        }
    }
    out[idx] = (_Float16)(acc * 0.25f);
}

// Final plain conv (12->3, 3x3, pad 1) + bias + tanh -> NCHW fp32 output.
// Input NHWC f16 with channel stride Cs (pad channels zero).
__global__ void k_to_rgb(const _Float16* __restrict__ act,
                         const float* __restrict__ w,
                         const float* __restrict__ bias,
                         float* __restrict__ out,
                         int Bn, int H, int W, int Cin, int Cs)
{
    long idx = (long)blockIdx.x * blockDim.x + threadIdx.x;
    long tot = (long)Bn * 3 * H * W;
    if (idx >= tot) return;
    int x = (int)(idx % W); long t = idx / W;
    int y = (int)(t % H); t /= H;
    int c = (int)(t % 3);
    int b = (int)(t / 3);
    float acc = bias[c];
    for (int ky = 0; ky < 3; ++ky) {
        int iy = y - 1 + ky;
        if (iy < 0 || iy >= H) continue;
        for (int kx = 0; kx < 3; ++kx) {
            int ix = x - 1 + kx;
            if (ix < 0 || ix >= W) continue;
            const _Float16* ap = act + ((long)((b * H + iy) * W) + ix) * Cs;
            for (int ci = 0; ci < Cin; ++ci)
                acc += (float)ap[ci] * w[((c * Cin + ci) * 3 + ky) * 3 + kx];
        }
    }
    out[idx] = tanhf(acc);
}

// ---------------------------------------------------------------------------
// Host orchestration
// ---------------------------------------------------------------------------
struct LSpec { int cin, coutc, k; };
static const LSpec SPEC[25] = {
    {3, 48, 5},  {24, 48, 3}, {24, 96, 3}, {48, 96, 3},  {48, 192, 3},
    {96, 192, 3},{96, 192, 3},{96, 192, 3},{96, 192, 3}, {96, 192, 3},
    {3, 48, 5},  {24, 48, 3}, {24, 96, 3}, {48, 192, 3}, {96, 192, 3},
    {96, 192, 3},{96, 192, 3},{96, 192, 3},{192, 192, 3},{96, 192, 3},
    {96, 96, 3}, {48, 96, 3}, {48, 48, 3}, {24, 24, 3},  {12, 3, 3}};

static inline unsigned cdiv(long a, long b) { return (unsigned)((a + b - 1) / b); }
static inline int pad32(int c) { return (c + 31) & ~31; }

extern "C" void kernel_launch(void* const* d_in, const int* in_sizes, int n_in,
                              void* d_out, int out_size, void* d_ws, size_t ws_size,
                              hipStream_t stream)
{
    (void)in_sizes; (void)n_in; (void)out_size; (void)ws_size;
    const float* x_in    = (const float*)d_in[0];
    const float* mask_in = (const float*)d_in[1];

    char* base = (char*)d_ws;
    size_t off = 0;
    auto alloc = [&](size_t bytes) -> char* {
        char* p = base + off;
        off = (off + bytes + 255) & ~(size_t)255;
        return p;
    };
    _Float16* xf16   = (_Float16*)alloc(262144L * 32 * 2);   // 3ch -> stride 32
    _Float16* ping   = (_Float16*)alloc(262144L * 64 * 2);   // worst: 48ch -> 64
    _Float16* pong   = (_Float16*)alloc(262144L * 64 * 2);
    _Float16* cat    = (_Float16*)alloc(16384L * 192 * 2);
    _Float16* x2buf  = (_Float16*)alloc(16384L * 96 * 2);
    _Float16* caout  = (_Float16*)alloc(16384L * 96 * 2);
    float*    convout= (float*)   alloc(262144L * 48 * 4);
    _Float16* Btw    = (_Float16*)alloc(1048576);
    _Float16* Btca   = (_Float16*)alloc(4096L * 864 * 2);
    _Float16* raww   = (_Float16*)alloc(4L * 16 * 1024 * 96 * 2);
    float*    mmb    = (float*)   alloc(4L * 1024 * 4);
    _Float16* bdbuf  = (_Float16*)alloc(4L * 1024 * 96 * 2);
    float*    f1     = (float*)   alloc(4L * 1024 * 1024 * 4);
    float*    f2     = (float*)   alloc(4L * 1024 * 1024 * 4);
    float*    corr   = convout;   // convout idle during attention
    float*    attn   = f1;        // f1 dead after fuse2

    auto gconv = [&](const _Float16* in, int Hin, int Win, int li,
                     int stride, int pad, int dil, int relu,
                     _Float16* outp, int outStride, int outOff, int cOutPad)
    {
        const int cin = SPEC[li].cin, coutc = SPEC[li].coutc, kk = SPEC[li].k;
        const int cnum = coutc / 2;
        const int cpad = pad32(cin);          // == input channel stride
        const int Kpad = kk * kk * cpad;
        const int Npad = pad32(coutc);
        const int Hout = (Hin + 2 * pad - dil * (kk - 1) - 1) / stride + 1;
        const int Wout = (Win + 2 * pad - dil * (kk - 1) - 1) / stride + 1;
        const long M = 4L * Hout * Wout;
        const float* w    = (const float*)d_in[2 + 2 * li];
        const float* bias = (const float*)d_in[3 + 2 * li];
        long wtot = (long)Npad * Kpad;
        k_pack_w<<<cdiv(wtot, 256), 256, 0, stream>>>(
            w, Btw, coutc, cin, kk, cpad, Kpad, wtot);
        dim3 g((unsigned)((M + 127) / 128), (unsigned)(Npad / 32));
        k_gemm<<<g, 128, 0, stream>>>(in, Btw, convout, Hin, Win, cpad,
                                      Hout, Wout, (int)M, coutc, Kpad, kk,
                                      stride, pad, dil);
        long atot = M * cOutPad;
        k_gated_act<<<cdiv(atot, 256), 256, 0, stream>>>(
            convout, bias, outp, M, cnum, coutc, cOutPad, outStride, outOff,
            relu);
    };

    // input NCHW fp32 -> NHWC f16 (stride 32, zero padded)
    k_nchw_to_nhwc<<<cdiv(262144L * 32, 256), 256, 0, stream>>>(
        x_in, xf16, 4, 3, 32, 256, 256);

    // ---- branch 1 (dilated conv tower) ----  (out: cnum -> padded stride)
    gconv(xf16, 256, 256, 0, 1, 2, 1, 0, ping, 32, 0, 32);      // 24ch @256
    gconv(ping, 256, 256, 1, 2, 1, 1, 0, pong, 32, 0, 32);      // 24ch @128
    gconv(pong, 128, 128, 2, 1, 1, 1, 0, ping, 64, 0, 64);      // 48ch @128
    gconv(ping, 128, 128, 3, 2, 1, 1, 0, pong, 64, 0, 64);      // 48ch @64
    gconv(pong, 64, 64, 4, 1, 1, 1, 0, ping, 96, 0, 96);        // 96ch @64
    gconv(ping, 64, 64, 5, 1, 1, 1, 0, pong, 96, 0, 96);
    gconv(pong, 64, 64, 6, 1, 2, 2, 0, ping, 96, 0, 96);
    gconv(ping, 64, 64, 7, 1, 4, 4, 0, pong, 96, 0, 96);
    gconv(pong, 64, 64, 8, 1, 8, 8, 0, ping, 96, 0, 96);
    gconv(ping, 64, 64, 9, 1, 16, 16, 0, cat, 192, 0, 96);      // concat[:,0:96]

    // ---- branch 2 (attention branch) ----
    gconv(xf16, 256, 256, 10, 1, 2, 1, 0, ping, 32, 0, 32);     // 24ch @256
    gconv(ping, 256, 256, 11, 2, 1, 1, 0, pong, 32, 0, 32);     // 24ch @128
    gconv(pong, 128, 128, 12, 1, 1, 1, 0, ping, 64, 0, 64);     // 48ch @128
    gconv(ping, 128, 128, 13, 2, 1, 1, 0, pong, 96, 0, 96);     // 96ch @64
    gconv(pong, 64, 64, 14, 1, 1, 1, 0, ping, 96, 0, 96);
    gconv(ping, 64, 64, 15, 1, 1, 1, 1 /*relu*/, x2buf, 96, 0, 96);

    // ---- contextual attention ----
    k_down2<<<cdiv(4L * 32 * 32 * 96, 256), 256, 0, stream>>>(
        x2buf, bdbuf, 4, 64, 64, 96);
    k_ca_wn<<<4 * 1024, 32, 0, stream>>>(bdbuf, Btca, 32, 32, 96, 864);
    k_mask<<<cdiv(4L * 1024, 256), 256, 0, stream>>>(
        mask_in, mmb, 4, 32, 32, 256, 256, 8);
    for (int b = 0; b < 4; ++b) {
        dim3 g(8, 32);    // M=1024 pixels x N=1024 patches, K=864 (WMMA GEMM)
        k_gemm<<<g, 128, 0, stream>>>(
            bdbuf + (long)b * 32 * 32 * 96, Btca + (long)b * 1024 * 864,
            corr + (long)b * 1024 * 1024,
            32, 32, 96, 32, 32, 1024, 1024, 864, 3, 1, 1, 1);
    }
    k_fuse1<<<cdiv(4L * 1024 * 1024, 256), 256, 0, stream>>>(corr, f1);
    k_fuse2<<<cdiv(4L * 1024 * 1024, 256), 256, 0, stream>>>(f1, f2);
    k_softmax<<<4 * 1024, 128, 0, stream>>>(f2, mmb, attn);
    k_raww<<<cdiv(4L * 16 * 1024 * 96, 256), 256, 0, stream>>>(x2buf, raww, 64, 96);
    k_deconv<<<cdiv(4L * 64 * 64 * 96, 256), 256, 0, stream>>>(attn, raww, caout, 96);

    gconv(caout, 64, 64, 16, 1, 1, 1, 0, ping, 96, 0, 96);
    gconv(ping, 64, 64, 17, 1, 1, 1, 0, cat, 192, 96, 96);      // concat[:,96:]

    // ---- decoder ----
    gconv(cat, 64, 64, 18, 1, 1, 1, 0, ping, 96, 0, 96);
    gconv(ping, 64, 64, 19, 1, 1, 1, 0, pong, 96, 0, 96);
    k_up2x<<<cdiv(4L * 128 * 128 * 96, 256), 256, 0, stream>>>(
        pong, ping, 4, 64, 64, 96);
    gconv(ping, 128, 128, 20, 1, 1, 1, 0, pong, 64, 0, 64);     // 48ch @128
    gconv(pong, 128, 128, 21, 1, 1, 1, 0, ping, 64, 0, 64);     // 48ch @128
    k_up2x<<<cdiv(4L * 256 * 256 * 64, 256), 256, 0, stream>>>(
        ping, pong, 4, 128, 128, 64);
    gconv(pong, 256, 256, 22, 1, 1, 1, 0, ping, 32, 0, 32);     // 24ch @256
    gconv(ping, 256, 256, 23, 1, 1, 1, 0, pong, 32, 0, 32);     // 12ch @256
    k_to_rgb<<<cdiv(4L * 3 * 256 * 256, 256), 256, 0, stream>>>(
        pong, (const float*)d_in[2 + 2 * 24], (const float*)d_in[3 + 2 * 24],
        (float*)d_out, 4, 256, 256, 12, 32);
}
